// FusedAttention_34076270526924
// MI455X (gfx1250) — matrix operand order repaired and checked by reading, not compiled, not run on hardware
//
#include <hip/hip_runtime.h>
#include <hip/hip_bf16.h>

typedef __attribute__((ext_vector_type(16))) _Float16 v16h;
typedef __attribute__((ext_vector_type(8)))  float    v8f;

union F16x16 { v16h v; uint4 q[2]; };

static __device__ __forceinline__ v8f wmma_f16(v16h a, v16h b, v8f c) {
    // (neg_a, A, neg_b, B, c_mod, C, reuse_a, reuse_b)
    return __builtin_amdgcn_wmma_f32_16x16x32_f16(false, a, false, b, (short)0, c, false, false);
}

// ---- CDNA5 async global->LDS copy (ASYNCcnt path) --------------------------
#if defined(__has_builtin)
#if __has_builtin(__builtin_amdgcn_global_load_async_to_lds_b128)
#define ASYNC_BUILTIN 1
#endif
#if __has_builtin(__builtin_amdgcn_s_wait_asynccnt)
#define WAIT_ASYNC_BUILTIN 1
#endif
#endif

typedef int i4v __attribute__((vector_size(16)));          // 128-bit payload
typedef __attribute__((address_space(1))) i4v* gi4p;       // global
typedef __attribute__((address_space(3))) i4v* li4p;       // LDS
typedef __attribute__((address_space(3))) void LA;

static __device__ __forceinline__ void async_cp16(void* lds, const void* g) {
#if defined(ASYNC_BUILTIN)
    __builtin_amdgcn_global_load_async_to_lds_b128((gi4p)g, (li4p)lds, 0, 0);
#else
    unsigned loff = (unsigned)(size_t)(LA*)lds;
    asm volatile("global_load_async_to_lds_b128 %0, %1, off"
                 :: "v"(loff), "v"(g) : "memory");
#endif
}
static __device__ __forceinline__ void wait_async0() {
#if defined(WAIT_ASYNC_BUILTIN)
    __builtin_amdgcn_s_wait_asynccnt(0);
#else
    asm volatile("s_wait_asynccnt 0" ::: "memory");
#endif
}

#define BB 4
#define SS 2048
#define HH 1024
#define NHD 16
#define HDI 64
#define MM (BB*SS)      // 8192
#define N3 (3*HH)       // 3072

// ---------------------------------------------------------------------------
// Kernel 0: convert + transpose weights to f16, N-major (so B tiles are
// contiguous along K for WMMA B-fragment loads).
// ---------------------------------------------------------------------------
__global__ __launch_bounds__(256) void k_convert(const float* __restrict__ aw,
                                                 const float* __restrict__ pw,
                                                 _Float16* __restrict__ wqkvT,
                                                 _Float16* __restrict__ projT) {
    const int total1 = N3 * HH;
    const int total2 = HH * HH;
    const int stride = gridDim.x * blockDim.x;
    for (int i = blockIdx.x * blockDim.x + threadIdx.x; i < total1; i += stride) {
        int n = i >> 10, k = i & 1023;                 // wqkvT[n][k] = aw[k][n]
        wqkvT[i] = (_Float16)aw[(size_t)k * N3 + n];
    }
    for (int i = blockIdx.x * blockDim.x + threadIdx.x; i < total2; i += stride) {
        int n = i >> 10, k = i & 1023;                 // projT[n][k] = pw[k][n]
        projT[i] = (_Float16)pw[(size_t)k * HH + n];
    }
}

// ---------------------------------------------------------------------------
// Kernel 1: LayerNorm (one row of H=1024 per block of 256 threads) -> f16
// ---------------------------------------------------------------------------
__global__ __launch_bounds__(256) void k_layernorm(const float* __restrict__ x,
                                                   const float* __restrict__ lw,
                                                   const float* __restrict__ lb,
                                                   _Float16* __restrict__ xh) {
    __shared__ float r1[256];
    __shared__ float r2[256];
    const int row = blockIdx.x;
    const int t = threadIdx.x;
    const float4 v = ((const float4*)(x + (size_t)row * HH))[t];
    float s  = v.x + v.y + v.z + v.w;
    float ss = v.x * v.x + v.y * v.y + v.z * v.z + v.w * v.w;
    r1[t] = s; r2[t] = ss;
    __syncthreads();
    for (int off = 128; off > 0; off >>= 1) {
        if (t < off) { r1[t] += r1[t + off]; r2[t] += r2[t + off]; }
        __syncthreads();
    }
    const float mu  = r1[0] * (1.0f / HH);
    const float var = r2[0] * (1.0f / HH) - mu * mu;
    const float rs  = rsqrtf(var + 1e-5f);
    const int c = t * 4;
    const float vv[4] = { v.x, v.y, v.z, v.w };
#pragma unroll
    for (int j = 0; j < 4; ++j)
        xh[(size_t)row * HH + c + j] = (_Float16)((vv[j] - mu) * rs * lw[c + j] + lb[c + j]);
}

// ---------------------------------------------------------------------------
// Shared GEMM core: C[128x128] += A[128xK] * B[Kx128], A row-major f16,
// BT stored N-major (BT[n][k]).  256 threads = 8 waves in a 2x4 grid, each
// wave computes a 64x32 patch as 4x2 WMMA 16x16 accumulators.
// Double-buffered LDS fed by async global->LDS copies (ASYNCcnt).
// Alds/Blds are [2*128][40] (buffer b at rows b*128..b*128+127).
// ---------------------------------------------------------------------------
__device__ __forceinline__ void gemm_core(const _Float16* __restrict__ A,
                                          const _Float16* __restrict__ BT,
                                          int mBase, int nBase, int K,
                                          _Float16 (*Alds)[40], _Float16 (*Blds)[40],
                                          v8f acc[4][2]) {
    const int tid  = threadIdx.x;
    const int lane = tid & 31, wid = tid >> 5;
    const int wm = wid >> 2, wn = wid & 3;
    const int lh = lane >> 4, ln15 = lane & 15;

    const v8f z = {};
#pragma unroll
    for (int i = 0; i < 4; ++i)
#pragma unroll
        for (int j = 0; j < 2; ++j) acc[i][j] = z;

    const int lrow = tid >> 1, lseg = tid & 1;   // 128 rows x two 16-half segments
    const _Float16* gA = A  + (size_t)(mBase + lrow) * K + lseg * 16;
    const _Float16* gB = BT + (size_t)(nBase + lrow) * K + lseg * 16;

    // prologue: tile 0 -> buffer 0
    async_cp16(&Alds[lrow][lseg * 16],     gA);
    async_cp16(&Alds[lrow][lseg * 16 + 8], gA + 8);
    async_cp16(&Blds[lrow][lseg * 16],     gB);
    async_cp16(&Blds[lrow][lseg * 16 + 8], gB + 8);

    int cur = 0;
    for (int k0 = 0; k0 < K; k0 += 32) {
        wait_async0();          // this wave's copies for buffer `cur` done
        __syncthreads();        // all waves' copies done; prev reads of `nxt` done

        if (k0 + 32 < K) {      // prefetch next tile into the other buffer
            const int nr = (cur ^ 1) * 128 + lrow;
            async_cp16(&Alds[nr][lseg * 16],     gA + k0 + 32);
            async_cp16(&Alds[nr][lseg * 16 + 8], gA + k0 + 40);
            async_cp16(&Blds[nr][lseg * 16],     gB + k0 + 32);
            async_cp16(&Blds[nr][lseg * 16 + 8], gB + k0 + 40);
        }

        const int rb = cur * 128;
        F16x16 af[4], bf[2];
#pragma unroll
        for (int i = 0; i < 4; ++i) {             // A 16x32: e0-7 -> K=h*8+e, e8-15 -> K=16+h*8+e
            const int r = rb + wm * 64 + i * 16 + ln15;
            af[i].q[0] = *(const uint4*)&Alds[r][lh * 8];
            af[i].q[1] = *(const uint4*)&Alds[r][16 + lh * 8];
        }
#pragma unroll
        for (int j = 0; j < 2; ++j) {             // B 32x16: lane=N, e0-7 -> K=h*8+e, e8-15 -> K=16+h*8+e
            const int r = rb + wn * 32 + j * 16 + ln15;
            bf[j].q[0] = *(const uint4*)&Blds[r][lh * 8];
            bf[j].q[1] = *(const uint4*)&Blds[r][16 + lh * 8];
        }
#pragma unroll
        for (int i = 0; i < 4; ++i)
#pragma unroll
            for (int j = 0; j < 2; ++j)
                acc[i][j] = wmma_f16(af[i].v, bf[j].v, acc[i][j]);

        cur ^= 1;
    }
}

// ---------------------------------------------------------------------------
// Kernel 2: QKV GEMM  (M=8192, N=3072, K=1024) + bias, Q pre-scaled 1/8,
// Q/K row-major per head, V stored transposed [d][s] per head.
// ---------------------------------------------------------------------------
__global__ __launch_bounds__(256) void k_qkv(const _Float16* __restrict__ xh,
                                             const _Float16* __restrict__ wqkvT,
                                             const float* __restrict__ attn_b,
                                             _Float16* __restrict__ qb,
                                             _Float16* __restrict__ kbuf,
                                             _Float16* __restrict__ vtb) {
    __shared__ alignas(16) _Float16 Alds[256][40];
    __shared__ alignas(16) _Float16 Blds[256][40];
    v8f acc[4][2];
    const int mBase = blockIdx.y * 128, nBase = blockIdx.x * 128;
    gemm_core(xh, wqkvT, mBase, nBase, HH, Alds, Blds, acc);

    const int tid = threadIdx.x, lane = tid & 31, wid = tid >> 5;
    const int wm = wid >> 2, wn = wid & 3, lh = lane >> 4, ln15 = lane & 15;
#pragma unroll
    for (int i = 0; i < 4; ++i)
#pragma unroll
        for (int j = 0; j < 2; ++j)
#pragma unroll
            for (int e = 0; e < 8; ++e) {         // C/D: M = e + 8h, N = lane&15
                const int row = mBase + wm * 64 + i * 16 + e + 8 * lh;
                const int col = nBase + wn * 32 + j * 16 + ln15;
                const float v = acc[i][j][e] + attn_b[col];
                const int sect = col >> 10, cc = col & 1023;
                const int head = cc >> 6, d = cc & 63;
                const int b = row >> 11, s = row & 2047;
                const int bh = b * NHD + head;
                if (sect == 0)
                    qb[((size_t)bh * SS + s) * HDI + d] = (_Float16)(v * 0.125f);
                else if (sect == 1)
                    kbuf[((size_t)bh * SS + s) * HDI + d] = (_Float16)v;
                else
                    vtb[((size_t)bh * HDI + d) * SS + s] = (_Float16)v;
            }
}

// ---------------------------------------------------------------------------
// Kernel 3: causal flash attention.  Block = (q-tile of 128, b*head);
// each of 8 waves owns 16 query rows; 64-key tiles staged in LDS via
// async global->LDS copies.
// ---------------------------------------------------------------------------
__global__ __launch_bounds__(256) void k_attn(const _Float16* __restrict__ qb,
                                              const _Float16* __restrict__ kbuf,
                                              const _Float16* __restrict__ vtb,
                                              _Float16* __restrict__ ctx) {
    __shared__ alignas(16) _Float16 Klds[64][72];
    __shared__ alignas(16) _Float16 Vlds[64][72];       // V^T tile: [d][key]
    __shared__ alignas(16) _Float16 Plds[8][16][72];    // per-wave P staging
    const int bh = blockIdx.y, qt = blockIdx.x;
    const int tid = threadIdx.x, lane = tid & 31, wid = tid >> 5;
    const int lh = lane >> 4, ln15 = lane & 15;
    const int qrow0 = qt * 128 + wid * 16;

    const _Float16* qbase = qb   + (size_t)bh * SS * HDI;
    const _Float16* kbase = kbuf + (size_t)bh * SS * HDI;
    const _Float16* vbase = vtb  + (size_t)bh * HDI * SS;

    F16x16 aQ[2];                                       // Q 16x64 as two 16x32 A-frags
    {
        const _Float16* qr = qbase + (size_t)(qrow0 + ln15) * HDI;
#pragma unroll
        for (int f = 0; f < 2; ++f) {
            aQ[f].q[0] = *(const uint4*)(qr + f * 32 + lh * 8);
            aQ[f].q[1] = *(const uint4*)(qr + f * 32 + 16 + lh * 8);
        }
    }

    float m_i[8], l_i[8];
    v8f o_acc[4];
    const v8f z = {};
#pragma unroll
    for (int e = 0; e < 8; ++e) { m_i[e] = -1e30f; l_i[e] = 0.0f; }
#pragma unroll
    for (int dn = 0; dn < 4; ++dn) o_acc[dn] = z;

    const int nKT  = qt * 2 + 2;                        // causal bound (uniform per block)
    const int lrow = tid >> 2, lseg = tid & 3;

    for (int kt = 0; kt < nKT; ++kt) {
        const int kb0 = kt * 64;
        {   // async stage K and V^T tiles (prev-iter LDS reads completed
            // before the trailing barrier of the previous iteration)
            const _Float16* gK = kbase + (size_t)(kb0 + lrow) * HDI + lseg * 16;
            const _Float16* gV = vbase + (size_t)lrow * SS + kb0 + lseg * 16;
            async_cp16(&Klds[lrow][lseg * 16],     gK);
            async_cp16(&Klds[lrow][lseg * 16 + 8], gK + 8);
            async_cp16(&Vlds[lrow][lseg * 16],     gV);
            async_cp16(&Vlds[lrow][lseg * 16 + 8], gV + 8);
        }
        wait_async0();
        __syncthreads();

        // ---- scores: S = Q * K^T  (K tile rows are exactly the B layout) ----
        v8f sArr[4];
#pragma unroll
        for (int nt = 0; nt < 4; ++nt) {
            v8f sa = z;
#pragma unroll
            for (int f = 0; f < 2; ++f) {
                F16x16 bk;
                const _Float16* p = &Klds[nt * 16 + ln15][f * 32 + lh * 8];
                bk.q[0] = *(const uint4*)p;
                bk.q[1] = *(const uint4*)(p + 16);
                sa = wmma_f16(aQ[f].v, bk.v, sa);
            }
            sArr[nt] = sa;
        }

        // ---- causal mask + online softmax (row reductions across 16-lane halves) ----
#pragma unroll
        for (int e = 0; e < 8; ++e) {
            const int qrow = qrow0 + e + 8 * lh;
            float mx = -1e30f;
#pragma unroll
            for (int nt = 0; nt < 4; ++nt) {
                const int col = kb0 + nt * 16 + ln15;
                float sv = sArr[nt][e];
                sv = (col <= qrow) ? sv : -1e30f;
                sArr[nt][e] = sv;
                mx = fmaxf(mx, sv);
            }
#pragma unroll
            for (int msk = 1; msk < 16; msk <<= 1) mx = fmaxf(mx, __shfl_xor(mx, msk, 32));
            const float nm = fmaxf(m_i[e], mx);
            const float alpha = __expf(m_i[e] - nm);
            float ps = 0.0f;
#pragma unroll
            for (int nt = 0; nt < 4; ++nt) {
                const float p = __expf(sArr[nt][e] - nm);
                sArr[nt][e] = p;
                ps += p;
            }
#pragma unroll
            for (int msk = 1; msk < 16; msk <<= 1) ps += __shfl_xor(ps, msk, 32);
            m_i[e] = nm;
            l_i[e] = l_i[e] * alpha + ps;
#pragma unroll
            for (int dn = 0; dn < 4; ++dn) o_acc[dn][e] *= alpha;
        }

        // ---- stage P as f16 in LDS (C layout -> A layout via LDS) ----
#pragma unroll
        for (int nt = 0; nt < 4; ++nt)
#pragma unroll
            for (int e = 0; e < 8; ++e)
                Plds[wid][e + 8 * lh][nt * 16 + ln15] = (_Float16)sArr[nt][e];
        __syncthreads();

        // ---- ctx: O += P * V  (V^T tile rows are exactly the B layout) ----
        F16x16 aP[2];
#pragma unroll
        for (int f = 0; f < 2; ++f) {
            const _Float16* pr = &Plds[wid][ln15][f * 32 + lh * 8];
            aP[f].q[0] = *(const uint4*)pr;
            aP[f].q[1] = *(const uint4*)(pr + 16);
        }
#pragma unroll
        for (int dn = 0; dn < 4; ++dn)
#pragma unroll
            for (int f = 0; f < 2; ++f) {
                F16x16 bv;
                const _Float16* p = &Vlds[dn * 16 + ln15][f * 32 + lh * 8];
                bv.q[0] = *(const uint4*)p;
                bv.q[1] = *(const uint4*)(p + 16);
                o_acc[dn] = wmma_f16(aP[f].v, bv.v, o_acc[dn]);
            }
        __syncthreads();
    }

    // ---- normalize + write ctx[b,s,h] in f16 ----
    const int b = bh >> 4, head = bh & 15;
#pragma unroll
    for (int dn = 0; dn < 4; ++dn)
#pragma unroll
        for (int e = 0; e < 8; ++e) {
            const int row = qrow0 + e + 8 * lh;
            const float val = o_acc[dn][e] / l_i[e];
            ctx[((size_t)b * SS + row) * HH + head * HDI + dn * 16 + ln15] = (_Float16)val;
        }
}

// ---------------------------------------------------------------------------
// Kernel 4: output projection (M=8192, N=1024, K=1024) + bias + residual
// ---------------------------------------------------------------------------
__global__ __launch_bounds__(256) void k_proj(const _Float16* __restrict__ ctx,
                                              const _Float16* __restrict__ projT,
                                              const float* __restrict__ proj_b,
                                              const float* __restrict__ xres,
                                              float* __restrict__ out) {
    __shared__ alignas(16) _Float16 Alds[256][40];
    __shared__ alignas(16) _Float16 Blds[256][40];
    v8f acc[4][2];
    const int mBase = blockIdx.y * 128, nBase = blockIdx.x * 128;
    gemm_core(ctx, projT, mBase, nBase, HH, Alds, Blds, acc);

    const int tid = threadIdx.x, lane = tid & 31, wid = tid >> 5;
    const int wm = wid >> 2, wn = wid & 3, lh = lane >> 4, ln15 = lane & 15;
#pragma unroll
    for (int i = 0; i < 4; ++i)
#pragma unroll
        for (int j = 0; j < 2; ++j)
#pragma unroll
            for (int e = 0; e < 8; ++e) {
                const int row = mBase + wm * 64 + i * 16 + e + 8 * lh;
                const int col = nBase + wn * 32 + j * 16 + ln15;
                out[(size_t)row * HH + col] =
                    acc[i][j][e] + proj_b[col] + xres[(size_t)row * HH + col];
            }
}

// ---------------------------------------------------------------------------
extern "C" void kernel_launch(void* const* d_in, const int* in_sizes, int n_in,
                              void* d_out, int out_size, void* d_ws, size_t ws_size,
                              hipStream_t stream) {
    (void)in_sizes; (void)n_in; (void)out_size; (void)ws_size;
    const float* x      = (const float*)d_in[0];
    const float* ln_w   = (const float*)d_in[1];
    const float* ln_b   = (const float*)d_in[2];
    const float* attn_w = (const float*)d_in[3];
    const float* attn_b = (const float*)d_in[4];
    const float* proj_w = (const float*)d_in[5];
    const float* proj_b = (const float*)d_in[6];
    float* out = (float*)d_out;

    char* w = (char*)d_ws;
    _Float16* xh    = (_Float16*)w; w += (size_t)MM * HH * 2;        // 16 MB
    _Float16* wqkvT = (_Float16*)w; w += (size_t)N3 * HH * 2;        //  6 MB
    _Float16* projT = (_Float16*)w; w += (size_t)HH * HH * 2;        //  2 MB
    _Float16* qb    = (_Float16*)w; w += (size_t)BB * NHD * SS * HDI * 2;   // 16.8 MB
    _Float16* kb    = (_Float16*)w; w += (size_t)BB * NHD * SS * HDI * 2;   // 16.8 MB
    _Float16* vtb   = (_Float16*)w; w += (size_t)BB * NHD * SS * HDI * 2;   // 16.8 MB
    _Float16* ctx   = (_Float16*)w; w += (size_t)MM * HH * 2;        // 16 MB

    k_convert<<<dim3(2048), dim3(256), 0, stream>>>(attn_w, proj_w, wqkvT, projT);
    k_layernorm<<<dim3(MM), dim3(256), 0, stream>>>(x, ln_w, ln_b, xh);
    k_qkv<<<dim3(N3 / 128, MM / 128), dim3(256), 0, stream>>>(xh, wqkvT, attn_b, qb, kb, vtb);
    k_attn<<<dim3(SS / 128, BB * NHD), dim3(256), 0, stream>>>(qb, kb, vtb, ctx);
    k_proj<<<dim3(HH / 128, MM / 128), dim3(256), 0, stream>>>(ctx, projT, proj_b, x, out);
}
